// SetAbstraction_43233140801874
// MI455X (gfx1250) — compile-verified, matrix operand order checked by reading, not run on hardware
//
#include <hip/hip_runtime.h>
#include <hip/hip_bf16.h>
#include <cstdint>
#include <cstddef>

// ---------------- problem constants ----------------
#define B_SZ   16
#define N_PTS  4096
#define D_FEAT 64
#define S_CTR  1024
#define K_NN   32
#define C_IN   67           // 3 + D_FEAT
#define C0P    96           // C_IN padded to multiple of 32
#define NPOS   (B_SZ * K_NN * S_CTR)   // 524288 positions (B,k,S)
#define BN_EPS 1e-5f

typedef __attribute__((ext_vector_type(16))) __bf16 v16bf;
typedef __attribute__((ext_vector_type(8)))  float  v8f;

__device__ __forceinline__ unsigned short f32_to_bf16_bits(float f) {
  union { float f; unsigned int u; } v; v.f = f;
  unsigned int r = v.u + 0x7FFFu + ((v.u >> 16) & 1u);   // round-to-nearest-even
  return (unsigned short)(r >> 16);
}
__device__ __forceinline__ float bf16_bits_to_f32(unsigned short h) {
  union { unsigned int u; float f; } v; v.u = ((unsigned int)h) << 16;
  return v.f;
}

// ---------------- 1) Farthest point sampling: one block per batch ----------------
__global__ __launch_bounds__(1024)
void fps_kernel(const float* __restrict__ xyz,   // [B,3,N]
                int* __restrict__ fps_idx,       // [B,S]
                float* __restrict__ new_xyz) {   // [B,S,3] row-major (s,3)
  const int b = blockIdx.x;
  const int t = threadIdx.x;
  __shared__ float sx[N_PTS], sy[N_PTS], sz[N_PTS];
  __shared__ float dist[N_PTS];
  __shared__ float red_d[1024];
  __shared__ int   red_i[1024];
  __shared__ int   s_far;

  const float* xb = xyz + (size_t)b * 3 * N_PTS;
  for (int i = t; i < N_PTS; i += 1024) {
    sx[i] = xb[i];
    sy[i] = xb[N_PTS + i];
    sz[i] = xb[2 * N_PTS + i];
    dist[i] = 1e10f;
  }
  if (t == 0) s_far = 0;
  __syncthreads();

  for (int s = 0; s < S_CTR; ++s) {
    const int far = s_far;
    const float cx = sx[far], cy = sy[far], cz = sz[far];
    if (t == 0) {
      fps_idx[b * S_CTR + s] = far;
      float* nz = new_xyz + ((size_t)b * S_CTR + s) * 3;
      nz[0] = cx; nz[1] = cy; nz[2] = cz;
    }
    float bestd = -1.0f; int besti = 0;
    #pragma unroll
    for (int i = t; i < N_PTS; i += 1024) {
      float dx = sx[i] - cx, dy = sy[i] - cy, dz = sz[i] - cz;
      float d  = dx * dx + dy * dy + dz * dz;
      float dm = fminf(dist[i], d);
      dist[i] = dm;
      if (dm > bestd) { bestd = dm; besti = i; }
    }
    red_d[t] = bestd; red_i[t] = besti;
    __syncthreads();
    for (int off = 512; off > 0; off >>= 1) {
      if (t < off) {
        float od = red_d[t + off]; int oi = red_i[t + off];
        if (od > red_d[t] || (od == red_d[t] && oi < red_i[t])) {
          red_d[t] = od; red_i[t] = oi;
        }
      }
      __syncthreads();
    }
    if (t == 0) s_far = red_i[0];
    __syncthreads();
  }
}

// ---------------- 2) KNN only at FPS centers: one thread per center ----------------
__global__ __launch_bounds__(256)
void knn_kernel(const float* __restrict__ xyz,     // [B,3,N]
                const int* __restrict__ fps_idx,   // [B,S]
                int* __restrict__ knn_idx) {       // [B,S,K]
  const int gid = blockIdx.x * 256 + threadIdx.x;  // 0 .. B*S-1 (block => same b)
  const int b = gid / S_CTR;
  __shared__ float sx[N_PTS], sy[N_PTS], sz[N_PTS];
  const float* xb = xyz + (size_t)b * 3 * N_PTS;
  for (int i = threadIdx.x; i < N_PTS; i += 256) {
    sx[i] = xb[i];
    sy[i] = xb[N_PTS + i];
    sz[i] = xb[2 * N_PTS + i];
  }
  __syncthreads();

  const int n0 = fps_idx[gid];
  const float cx = sx[n0], cy = sy[n0], cz = sz[n0];

  float bd[K_NN]; int bi[K_NN];
  #pragma unroll
  for (int j = 0; j < K_NN; ++j) { bd[j] = 3e38f; bi[j] = 0; }

  for (int m = 0; m < N_PTS; ++m) {
    if (m == n0) continue;                       // reference drops "self" column
    float dx = sx[m] - cx, dy = sy[m] - cy, dz = sz[m] - cz;
    float d  = dx * dx + dy * dy + dz * dz;
    if (d < bd[K_NN - 1]) {
      int j = K_NN - 1;
      while (j > 0 && bd[j - 1] > d) { bd[j] = bd[j - 1]; bi[j] = bi[j - 1]; --j; }
      bd[j] = d; bi[j] = m;
    }
  }
  int* o = knn_idx + (size_t)gid * K_NN;
  #pragma unroll
  for (int j = 0; j < K_NN; ++j) o[j] = bi[j];
}

// ---------------- 3) Gather + concat -> X0 [NPOS, 96] bf16 ----------------
__global__ __launch_bounds__(256)
void gather_kernel(const float* __restrict__ xyz,      // [B,3,N]
                   const float* __restrict__ points,   // [B,D,N]
                   const int* __restrict__ knn_idx,    // [B,S,K]
                   const float* __restrict__ new_xyz,  // [B,S,3]
                   unsigned short* __restrict__ X0) {  // [NPOS, C0P] bf16
  const int row = blockIdx.x * 256 + threadIdx.x;      // pos = ((b*K + kk)*S + s)
  if (row >= NPOS) return;
  const int b  = row / (K_NN * S_CTR);
  const int r2 = row % (K_NN * S_CTR);
  const int kk = r2 / S_CTR;
  const int s  = r2 % S_CTR;

  const int idx = knn_idx[((size_t)b * S_CTR + s) * K_NN + kk];
  const float* cen = new_xyz + ((size_t)b * S_CTR + s) * 3;
  const float* xb  = xyz + (size_t)b * 3 * N_PTS;

  unsigned short* o = X0 + (size_t)row * C0P;
  o[0] = f32_to_bf16_bits(xb[idx]              - cen[0]);
  o[1] = f32_to_bf16_bits(xb[N_PTS + idx]      - cen[1]);
  o[2] = f32_to_bf16_bits(xb[2 * N_PTS + idx]  - cen[2]);
  const float* pb = points + (size_t)b * D_FEAT * N_PTS;
  #pragma unroll 8
  for (int c = 0; c < D_FEAT; ++c)
    o[3 + c] = f32_to_bf16_bits(pb[(size_t)c * N_PTS + idx]);
  #pragma unroll
  for (int c = C_IN; c < C0P; ++c) o[c] = 0;   // zero pad (weights also zero-padded)
}

// ---------------- 4) convert + pad weights to bf16 ----------------
__global__ void wconv_kernel(const float* __restrict__ w, unsigned short* __restrict__ wbf,
                             int OC, int Cin, int KD) {
  const int i = blockIdx.x * 256 + threadIdx.x;
  if (i >= OC * KD) return;
  const int oc = i / KD, c = i % KD;
  wbf[i] = (c < Cin) ? f32_to_bf16_bits(w[(size_t)oc * Cin + c]) : (unsigned short)0;
}

// ---------------- 5) GEMM (WMMA bf16) + bias + BN-stats epilogue ----------------
// Y[pos, oc] = sum_c W[oc,c] * X[pos,c] + bias[oc]; stats[oc]+=v, stats[OC+oc]+=v*v
// Each block stages the whole weight matrix in LDS once, then sweeps NT N-tiles
// (NT*16 positions) to amortize the stage 8x vs one-tile blocks.
template <int OC, int KD, int NT>
__global__ __launch_bounds__(256)
void gemm_kernel(const unsigned short* __restrict__ X,   // [NPOS, KD] bf16
                 const unsigned short* __restrict__ W,   // [OC, KD]  bf16
                 const float* __restrict__ bias,         // [OC]
                 unsigned short* __restrict__ Y,         // [NPOS, OC] bf16
                 float* __restrict__ stats) {            // [2*OC]
  const int lane  = threadIdx.x;          // 0..31
  const int wave  = threadIdx.y;          // 0..OC/16-1 (M-tile per wave)
  const int tid   = wave * 32 + lane;
  const int nthr  = 32 * (OC / 16);

  __shared__ __align__(32) unsigned short sW[OC * KD];
  __shared__ float sSum[OC], sSq[OC];
  for (int i = tid; i < OC * KD; i += nthr) sW[i] = W[i];
  for (int i = tid; i < OC;      i += nthr) { sSum[i] = 0.f; sSq[i] = 0.f; }
  __syncthreads();

  const int mBase = wave * 16;
  const int l15   = lane & 15;
  const int khalf = (lane >> 4) * 16;
  const int mOff  = mBase + (lane >> 4) * 8;

  for (int nt = 0; nt < NT; ++nt) {
    const int nBase = (blockIdx.x * NT + nt) * 16;

    v8f acc = {};
    #pragma unroll
    for (int kb = 0; kb < KD; kb += 32) {
      const int kOff = kb + khalf;
      // A fragment: 16x32 weight tile (rows = oc), contiguous 16 halfs / lane (LDS)
      v16bf a = *(const v16bf*)(&sW[(mBase + l15) * KD + kOff]);
      // B fragment: 32x16 activation tile (cols = pos), contiguous 16 halfs / lane
      const size_t xoff = (size_t)(nBase + l15) * KD + kOff;
      v16bf bf = *(const v16bf*)(&X[xoff]);
      __builtin_prefetch(&X[xoff + 16 * KD], 0, 1);        // global_prefetch_b8
      acc = __builtin_amdgcn_wmma_f32_16x16x32_bf16(
          /*neg_a=*/false, a, /*neg_b=*/false, bf,
          /*c_mod=*/(short)0, acc, /*reuse_a=*/false, /*reuse_b=*/false);
    }

    // C/D layout: VGPR r -> M = mBase + r + (lane/16)*8 ; N = lane%16
    const int n = nBase + l15;
    #pragma unroll
    for (int r = 0; r < 8; ++r) {
      const int oc = mOff + r;
      const float v = acc[r] + bias[oc];
      Y[(size_t)n * OC + oc] = f32_to_bf16_bits(v);
      atomicAdd(&sSum[oc], v);          // ds_add_f32
      atomicAdd(&sSq[oc],  v * v);
    }
  }
  __syncthreads();
  for (int i = tid; i < OC; i += nthr) {
    atomicAdd(&stats[i],       sSum[i]);   // global_atomic_add_f32
    atomicAdd(&stats[OC + i],  sSq[i]);
  }
}

// ---------------- 6) BN + ReLU elementwise -> next layer input (bf16) ----------------
__global__ __launch_bounds__(256)
void bn_relu_kernel(const unsigned short* __restrict__ Y, const float* __restrict__ stats,
                    const float* __restrict__ gamma, const float* __restrict__ beta,
                    unsigned short* __restrict__ Xn, int OC, size_t total) {
  const size_t i = (size_t)blockIdx.x * 256 + threadIdx.x;
  if (i >= total) return;
  const int oc = (int)(i % OC);
  const float cnt  = (float)NPOS;
  const float mean = stats[oc] / cnt;
  const float var  = stats[OC + oc] / cnt - mean * mean;
  const float sc   = rsqrtf(var + BN_EPS) * gamma[oc];
  const float v    = (bf16_bits_to_f32(Y[i]) - mean) * sc + beta[oc];
  Xn[i] = f32_to_bf16_bits(fmaxf(v, 0.0f));
}

// ---------------- 7) new_xyz transpose to output [B,3,S] ----------------
__global__ void xyz_out_kernel(const float* __restrict__ new_xyz, float* __restrict__ out) {
  const int gid = blockIdx.x * 256 + threadIdx.x;    // B*3*S
  if (gid >= B_SZ * 3 * S_CTR) return;
  const int b = gid / (3 * S_CTR);
  const int r = gid % (3 * S_CTR);
  const int c = r / S_CTR, s = r % S_CTR;
  out[gid] = new_xyz[((size_t)b * S_CTR + s) * 3 + c];
}

// ---------------- 8) last-layer BN + ReLU + max over k -> new_points ----------------
__global__ __launch_bounds__(256)
void final_max_kernel(const unsigned short* __restrict__ Y3, const float* __restrict__ stats,
                      const float* __restrict__ gamma, const float* __restrict__ beta,
                      float* __restrict__ out) {      // out base = d_out + B*3*S
  const int gid = blockIdx.x * 256 + threadIdx.x;     // B*128*S
  if (gid >= B_SZ * 128 * S_CTR) return;
  const int b  = gid / (128 * S_CTR);
  const int r  = gid % (128 * S_CTR);
  const int oc = r / S_CTR, s = r % S_CTR;
  const float cnt  = (float)NPOS;
  const float mean = stats[oc] / cnt;
  const float var  = stats[128 + oc] / cnt - mean * mean;
  const float sc   = rsqrtf(var + BN_EPS) * gamma[oc];
  const float sh   = beta[oc] - mean * sc;
  float m = 0.0f;                                     // post-ReLU values are >= 0
  #pragma unroll 4
  for (int kk = 0; kk < K_NN; ++kk) {
    const size_t pos = ((size_t)(b * K_NN + kk)) * S_CTR + s;
    const float v = fmaxf(bf16_bits_to_f32(Y3[pos * 128 + oc]) * sc + sh, 0.0f);
    m = fmaxf(m, v);
  }
  out[gid] = m;
}

// ---------------- host orchestration ----------------
extern "C" void kernel_launch(void* const* d_in, const int* in_sizes, int n_in,
                              void* d_out, int out_size, void* d_ws, size_t ws_size,
                              hipStream_t stream) {
  const float* xyz    = (const float*)d_in[0];   // [16,3,4096]
  const float* points = (const float*)d_in[1];   // [16,64,4096]
  const float* w1 = (const float*)d_in[2];       // [64,67]
  const float* w2 = (const float*)d_in[3];       // [64,64]
  const float* w3 = (const float*)d_in[4];       // [128,64]
  const float* b1 = (const float*)d_in[5];
  const float* b2 = (const float*)d_in[6];
  const float* b3 = (const float*)d_in[7];
  const float* g1 = (const float*)d_in[8];
  const float* g2 = (const float*)d_in[9];
  const float* g3 = (const float*)d_in[10];
  const float* t1 = (const float*)d_in[11];
  const float* t2 = (const float*)d_in[12];
  const float* t3 = (const float*)d_in[13];
  float* out = (float*)d_out;

  // workspace carve-up (256B aligned)
  char* ws = (char*)d_ws;
  size_t cur = 0;
  auto take = [&](size_t bytes) -> size_t {
    size_t o = cur; cur += (bytes + 255) & ~(size_t)255; return o;
  };
  int*   fps_idx = (int*)  (ws + take((size_t)B_SZ * S_CTR * 4));
  float* new_xyz = (float*)(ws + take((size_t)B_SZ * S_CTR * 3 * 4));
  int*   knn_idx = (int*)  (ws + take((size_t)B_SZ * S_CTR * K_NN * 4));
  float* stats   = (float*)(ws + take(3 * 256 * 4));              // 3 layers x [2*128]
  unsigned short* wbf1 = (unsigned short*)(ws + take((size_t)64  * C0P * 2));
  unsigned short* wbf2 = (unsigned short*)(ws + take((size_t)64  * 64  * 2));
  unsigned short* wbf3 = (unsigned short*)(ws + take((size_t)128 * 64  * 2));
  unsigned short* X0   = (unsigned short*)(ws + take((size_t)NPOS * C0P * 2)); // 96-ch
  unsigned short* X1   = (unsigned short*)(ws + take((size_t)NPOS * 64  * 2)); // 64-ch
  unsigned short* Y    = (unsigned short*)(ws + take((size_t)NPOS * 128 * 2)); // up to 128-ch
  (void)ws_size; (void)in_sizes; (void)n_in; (void)out_size;

  float* stats1 = stats;
  float* stats2 = stats + 256;
  float* stats3 = stats + 512;
  hipMemsetAsync(stats, 0, 3 * 256 * 4, stream);

  // weights -> bf16 (padded)
  wconv_kernel<<<(64 * C0P + 255) / 256, 256, 0, stream>>>(w1, wbf1, 64, C_IN, C0P);
  wconv_kernel<<<(64 * 64 + 255) / 256, 256, 0, stream>>>(w2, wbf2, 64, 64, 64);
  wconv_kernel<<<(128 * 64 + 255) / 256, 256, 0, stream>>>(w3, wbf3, 128, 64, 64);

  // geometry pipeline
  fps_kernel<<<B_SZ, 1024, 0, stream>>>(xyz, fps_idx, new_xyz);
  knn_kernel<<<(B_SZ * S_CTR) / 256, 256, 0, stream>>>(xyz, fps_idx, knn_idx);
  gather_kernel<<<(NPOS + 255) / 256, 256, 0, stream>>>(xyz, points, knn_idx, new_xyz, X0);

  constexpr int NT = 8;   // N-tiles per block (amortize LDS weight stage)

  // MLP: layer 1 (96 -> 64)
  gemm_kernel<64, C0P, NT><<<dim3(NPOS / (16 * NT)), dim3(32, 4), 0, stream>>>(
      X0, wbf1, b1, Y, stats1);
  bn_relu_kernel<<<(unsigned)(((size_t)NPOS * 64 + 255) / 256), 256, 0, stream>>>(
      Y, stats1, g1, t1, X1, 64, (size_t)NPOS * 64);

  // layer 2 (64 -> 64) ; reuse X0 buffer as 64-ch output
  gemm_kernel<64, 64, NT><<<dim3(NPOS / (16 * NT)), dim3(32, 4), 0, stream>>>(
      X1, wbf2, b2, Y, stats2);
  bn_relu_kernel<<<(unsigned)(((size_t)NPOS * 64 + 255) / 256), 256, 0, stream>>>(
      Y, stats2, g2, t2, X0, 64, (size_t)NPOS * 64);

  // layer 3 (64 -> 128)
  gemm_kernel<128, 64, NT><<<dim3(NPOS / (16 * NT)), dim3(32, 8), 0, stream>>>(
      X0, wbf3, b3, Y, stats3);

  // outputs
  xyz_out_kernel<<<(B_SZ * 3 * S_CTR + 255) / 256, 256, 0, stream>>>(new_xyz, out);
  final_max_kernel<<<(B_SZ * 128 * S_CTR + 255) / 256, 256, 0, stream>>>(
      Y, stats3, g3, t3, out + (size_t)B_SZ * 3 * S_CTR);
}
